// Block_84765474554515
// MI455X (gfx1250) — matrix operand order repaired and checked by reading, not compiled, not run on hardware
//
#include <hip/hip_runtime.h>
#include <hip/hip_bf16.h>
#include <math.h>

typedef __attribute__((ext_vector_type(16))) __bf16 v16bf;
typedef __attribute__((ext_vector_type(8)))  __bf16 v8bf;
typedef __attribute__((ext_vector_type(8)))  float  v8f;

// ---------------- WMMA helper (gfx1250: V_WMMA_F32_16X16X32_BF16) ------------
__device__ __forceinline__ v8f wmma_bf16f32(v16bf a, v16bf b, v8f c) {
#if defined(__HIP_DEVICE_COMPILE__)
  // 8 args: (neg_a, A, neg_b, B, c_mod, C, reuse_a, reuse_b)
  return __builtin_amdgcn_wmma_f32_16x16x32_bf16(false, a, false, b, (short)0, c, false, false);
#else
  (void)a; (void)b;
  return c;
#endif
}

// ---------------- CDNA5 async global->LDS (ASYNCcnt pipe) --------------------
// GLOBAL_LOAD_ASYNC_TO_LDS_B128: vdst = per-lane LDS byte address, vaddr = 64b
// global address.  Tracked by ASYNCcnt; we wait explicitly before the barrier.
__device__ __forceinline__ void async_load_b128(unsigned lds_addr, const void* gptr) {
#if defined(__HIP_DEVICE_COMPILE__)
  asm volatile("global_load_async_to_lds_b128 %0, %1, off"
               :: "v"(lds_addr), "v"(gptr) : "memory");
#else
  (void)lds_addr; (void)gptr;
#endif
}
__device__ __forceinline__ void wait_async0() {
#if defined(__HIP_DEVICE_COMPILE__)
  asm volatile("s_wait_asynccnt 0x0" ::: "memory");
#endif
}
// Low 32 bits of a generic pointer into the LDS aperture are the LDS address
// (ISA 10.2).
__device__ __forceinline__ unsigned lds_addr_of(const void* p) {
  return (unsigned)(uintptr_t)p;
}

// A-fragment (16x32 bf16, MxK): lanes 0-15 rows M=0..15 hold K 0..7 & 16..23,
// lanes 16-31 hold K 8..15 & 24..31 (CDNA5 ISA layout).
__device__ __forceinline__ v16bf frag_a(const __bf16* base, int stride, int lane) {
  const int row  = lane & 15;
  const int half = lane >> 4;
  const __bf16* p = base + (size_t)row * stride + half * 8;
  v8bf lo = *(const v8bf*)p;
  v8bf hi = *(const v8bf*)(p + 16);
  v16bf a;
#pragma unroll
  for (int i = 0; i < 8; ++i) { a[i] = lo[i]; a[8 + i] = hi[i]; }
  return a;
}

// B-fragment (32x16 bf16, KxN) read from a K-contiguous (transposed) tile:
// lane&15 selects column N, lanes 0-15 hold K 0..7 & 16..23, lanes 16-31 K 8..15 & 24..31.
__device__ __forceinline__ v16bf frag_b(const __bf16* base, int stride, int lane) {
  const int col  = lane & 15;
  const int koff = (lane >> 4) * 8;
  const __bf16* p = base + (size_t)col * stride + koff;
  v8bf lo = *(const v8bf*)p;
  v8bf hi = *(const v8bf*)(p + 16);
  v16bf b;
#pragma unroll
  for (int i = 0; i < 8; ++i) { b[i] = lo[i]; b[8 + i] = hi[i]; }
  return b;
}

// ---------------- elementwise cast ----------------
__global__ __launch_bounds__(256) void cast_bf16_kernel(const float* __restrict__ src,
                                                        __bf16* __restrict__ dst, size_t n) {
  size_t i = (size_t)blockIdx.x * 256 + threadIdx.x;
  const size_t stride = (size_t)gridDim.x * 256;
  for (; i < n; i += stride) dst[i] = (__bf16)src[i];
}

// ---------------- RMSNorm over C=1024 ----------------
template <bool WRITE_F32>
__global__ __launch_bounds__(256) void rmsnorm_kernel(const float* __restrict__ X,
                                                      __bf16* __restrict__ Hb,
                                                      float* __restrict__ Hf) {
  const int row = blockIdx.x;
  const float* xr = X + (size_t)row * 1024;
  float v[4];
  float ss = 0.f;
#pragma unroll
  for (int i = 0; i < 4; ++i) {
    v[i] = xr[threadIdx.x + i * 256];
    ss += v[i] * v[i];
  }
  __shared__ float sh[256];
  sh[threadIdx.x] = ss;
  __syncthreads();
  for (int s = 128; s > 0; s >>= 1) {
    if ((int)threadIdx.x < s) sh[threadIdx.x] += sh[threadIdx.x + s];
    __syncthreads();
  }
  const float inv = rsqrtf(sh[0] / 1024.0f + 1.1920929e-07f);
#pragma unroll
  for (int i = 0; i < 4; ++i) {
    const int c = threadIdx.x + i * 256;
    const float hv = v[i] * inv;
    Hb[(size_t)row * 1024 + c] = (__bf16)hv;
    if (WRITE_F32) Hf[(size_t)row * 1024 + c] = hv;
  }
}

// ---------------- generic WMMA GEMM: Out = A[bf16, MxK] * B[bf16, KxN] -------
// MODE 0: OutF = acc
// MODE 1: OutF = acc + R                         (residual fuse)
// MODE 2: OutB[(off+m)] = bf16(relu(acc)^2), A row = perm[off+m]   (per-expert)
// MODE 3: OutF[perm[off+m]] = acc + R[perm[off+m]], A row = off+m  (per-expert)
template <int MODE>
__global__ __launch_bounds__(256) void gemm_bf16_wmma(
    const __bf16* __restrict__ A, const __bf16* __restrict__ Bmat,
    const float* __restrict__ R, float* __restrict__ OutF, __bf16* __restrict__ OutB,
    const int* __restrict__ perm, const int* __restrict__ meta,
    int M, int N, int K) {
  __shared__ __align__(16) __bf16 As[128][40];  // padded: conflict-free b128 frag loads
  __shared__ __align__(16) __bf16 Bt[128][40];  // B tile stored K-contiguous (transposed)

  const int e = blockIdx.z;
  const int count = (MODE >= 2) ? meta[e] : M;
  const int off   = (MODE >= 2) ? meta[4 + e] : 0;
  const int m0 = blockIdx.y * 128;
  const int n0 = blockIdx.x * 128;
  if (m0 >= count) return;  // block-uniform early exit (expert may be small/empty)

  const __bf16* Bw = Bmat + (size_t)e * (size_t)K * (size_t)N;

  const int tid  = threadIdx.x;
  const int lane = tid & 31;
  const int wv   = tid >> 5;          // 8 waves
  const int m_base = (wv & 3) * 32;   // 4 wave-rows x 32
  const int n_base = (wv >> 2) * 64;  // 2 wave-cols x 64

  // cooperative loader assignments
  const int la_r   = tid >> 1;        // A: row within tile, 16 halves per thread
  const int la_seg = tid & 1;
  const int lb_k   = tid >> 3;        // B: K row, 16 halves along N per thread
  const int lb_seg = tid & 7;

  int mg = m0 + la_r;
  if (mg > count - 1) mg = count - 1;  // clamp; extra rows never stored
  int arow;
  if (MODE == 2)      arow = perm[off + mg];
  else if (MODE == 3) arow = off + mg;
  else                arow = mg;
  const __bf16* Asrc = A + (size_t)arow * (size_t)K + la_seg * 16;
  const __bf16* Bsrc = Bw + (size_t)lb_k * (size_t)N + n0 + lb_seg * 16;

  const unsigned as_dst0 = lds_addr_of(&As[la_r][la_seg * 16]);
  const unsigned as_dst1 = lds_addr_of(&As[la_r][la_seg * 16 + 8]);

  v8f acc[2][4];
#pragma unroll
  for (int i = 0; i < 2; ++i)
#pragma unroll
    for (int j = 0; j < 4; ++j)
#pragma unroll
      for (int r = 0; r < 8; ++r) acc[i][j][r] = 0.0f;

  for (int kk = 0; kk < K; kk += 32) {
    __syncthreads();
    {
      // A tile: CDNA5 async global->LDS copies (layout-preserving)
      async_load_b128(as_dst0, Asrc + kk);
      async_load_b128(as_dst1, Asrc + kk + 8);
      // B tile: register transpose into K-contiguous layout
      v8bf b0 = *(const v8bf*)(Bsrc + (size_t)kk * N);
      v8bf b1 = *(const v8bf*)(Bsrc + (size_t)kk * N + 8);
#pragma unroll
      for (int i = 0; i < 8; ++i) {
        Bt[lb_seg * 16 + i][lb_k]     = b0[i];
        Bt[lb_seg * 16 + 8 + i][lb_k] = b1[i];
      }
      if (kk + 32 < K) {  // CDNA5 global_prefetch_b8 path for the next K tile
        __builtin_prefetch(Asrc + kk + 32, 0, 1);
        __builtin_prefetch(Bsrc + (size_t)(kk + 32) * N, 0, 1);
      }
    }
    wait_async0();
    __syncthreads();

    v16bf a0 = frag_a(&As[m_base][0], 40, lane);
    v16bf a1 = frag_a(&As[m_base + 16][0], 40, lane);
#pragma unroll
    for (int j = 0; j < 4; ++j) {
      v16bf bb = frag_b(&Bt[n_base + j * 16][0], 40, lane);
      acc[0][j] = wmma_bf16f32(a0, bb, acc[0][j]);
      acc[1][j] = wmma_bf16f32(a1, bb, acc[1][j]);
    }
  }

  // C/D layout: VGPR r -> row r (lanes 0-15) / r+8 (lanes 16-31); col = lane&15
  const int half = lane >> 4;
  const int cn   = lane & 15;
#pragma unroll
  for (int i = 0; i < 2; ++i) {
#pragma unroll
    for (int j = 0; j < 4; ++j) {
#pragma unroll
      for (int r = 0; r < 8; ++r) {
        const int m = m0 + m_base + i * 16 + r + half * 8;
        const int n = n0 + n_base + j * 16 + cn;
        const float v = acc[i][j][r];
        if (MODE == 0) {
          OutF[(size_t)m * N + n] = v;
        } else if (MODE == 1) {
          OutF[(size_t)m * N + n] = v + R[(size_t)m * N + n];
        } else if (MODE == 2) {
          if (m < count) {
            const float rl = v > 0.0f ? v : 0.0f;
            OutB[(size_t)(off + m) * N + n] = (__bf16)(rl * rl);
          }
        } else {
          if (m < count) {
            const int tok = perm[off + m];
            OutF[(size_t)tok * N + n] = v + R[(size_t)tok * N + n];
          }
        }
      }
    }
  }
}

// ---------------- per-(token,head) RMSNorm + rotary, pack to [B,H,T,D] bf16 --
__global__ __launch_bounds__(256) void qk_prep_kernel(
    const float* __restrict__ qf, const float* __restrict__ kf, const float* __restrict__ vf,
    __bf16* __restrict__ Qh, __bf16* __restrict__ Kh, __bf16* __restrict__ Vh) {
  const int wave = threadIdx.x >> 5, lane = threadIdx.x & 31;
  const int wg = blockIdx.x * 8 + wave;     // N_TOK*H = 65536 work items
  const int h = wg & 15;
  const int n = wg >> 4;                    // token 0..4095
  const int t = n & 2047;                   // position within T
  const int b = n >> 11;
  const size_t src = (size_t)n * 1024 + h * 64;
  float q1 = qf[src + lane], q2 = qf[src + lane + 32];
  float k1 = kf[src + lane], k2 = kf[src + lane + 32];
  const float v1 = vf[src + lane], v2 = vf[src + lane + 32];
  float sq = q1 * q1 + q2 * q2;
  float sk = k1 * k1 + k2 * k2;
#pragma unroll
  for (int m = 16; m >= 1; m >>= 1) {
    sq += __shfl_xor(sq, m, 32);
    sk += __shfl_xor(sk, m, 32);
  }
  const float invq = rsqrtf(sq / 64.0f + 1.1920929e-07f);
  const float invk = rsqrtf(sk / 64.0f + 1.1920929e-07f);
  q1 *= invq; q2 *= invq; k1 *= invk; k2 *= invk;
  const float freq = (float)t * powf(10000.0f, -(float)lane / 32.0f);
  const float cs = cosf(freq), sn = sinf(freq);
  const size_t dst = (((size_t)(b * 16 + h)) * 2048 + t) * 64;
  Qh[dst + lane]      = (__bf16)(q1 * cs + q2 * sn);
  Qh[dst + lane + 32] = (__bf16)(-q1 * sn + q2 * cs);
  Kh[dst + lane]      = (__bf16)(k1 * cs + k2 * sn);
  Kh[dst + lane + 32] = (__bf16)(-k1 * sn + k2 * cs);
  Vh[dst + lane]      = (__bf16)v1;
  Vh[dst + lane + 32] = (__bf16)v2;
}

// ---------------- causal flash attention, WMMA S=QK^T and O+=PV --------------
__global__ __launch_bounds__(256) void flash_attn_kernel(
    const __bf16* __restrict__ Qh, const __bf16* __restrict__ Kh,
    const __bf16* __restrict__ Vh, __bf16* __restrict__ Yb) {
  __shared__ __align__(16) __bf16 Ks[32][72];       // [key][d], pad 72 -> conflict-free
  __shared__ __align__(16) __bf16 Vt[64][40];       // [d][key], transposed, pad 40
  __shared__ __align__(16) __bf16 Pa[8][16][40];    // per-wave P staging (A-frag layout)

  const int qt = blockIdx.x & 15;
  const int h  = (blockIdx.x >> 4) & 15;
  const int b  = blockIdx.x >> 8;
  const int wave = threadIdx.x >> 5, lane = threadIdx.x & 31;

  const size_t base = ((size_t)(b * 16 + h)) * 2048 * 64;
  const __bf16* Qb = Qh + base;
  const __bf16* Kb = Kh + base;
  const __bf16* Vb = Vh + base;

  const int q0 = qt * 128 + wave * 16;  // 16 q rows per wave
  const v16bf qa0 = frag_a(Qb + (size_t)q0 * 64 + 0, 64, lane);
  const v16bf qa1 = frag_a(Qb + (size_t)q0 * 64 + 32, 64, lane);

  v8f o[4];
  float mr[8], lr[8];
#pragma unroll
  for (int j = 0; j < 4; ++j)
#pragma unroll
    for (int r = 0; r < 8; ++r) o[j][r] = 0.0f;
#pragma unroll
  for (int r = 0; r < 8; ++r) { mr[r] = -3.0e38f; lr[r] = 0.0f; }

  const int half = lane >> 4;
  const int cn   = lane & 15;
  const int krow = threadIdx.x >> 3;
  const int seg  = threadIdx.x & 7;
  const unsigned ks_dst = lds_addr_of(&Ks[krow][seg * 8]);
  const int ntiles = qt * 4 + 4;  // causal: keys up to q-tile end

  for (int kt = 0; kt < ntiles; ++kt) {
    const int key0 = kt * 32;
    __syncthreads();
    {   // K tile: async global->LDS (layout preserved); V transposed manually
      async_load_b128(ks_dst, Kb + (size_t)(key0 + krow) * 64 + seg * 8);
      v8bf vv = *(const v8bf*)(Vb + (size_t)(key0 + krow) * 64 + seg * 8);
#pragma unroll
      for (int i = 0; i < 8; ++i) Vt[seg * 8 + i][krow] = vv[i];
    }
    wait_async0();
    __syncthreads();

    // S = Q * K^T  (two 16-key N-frags, D=64 -> two K-steps each)
    v8f s0, s1;
#pragma unroll
    for (int r = 0; r < 8; ++r) { s0[r] = 0.0f; s1[r] = 0.0f; }
    {
      v16bf kb00 = frag_b(&Ks[0][0], 72, lane);
      v16bf kb01 = frag_b(&Ks[0][32], 72, lane);
      s0 = wmma_bf16f32(qa0, kb00, s0);
      s0 = wmma_bf16f32(qa1, kb01, s0);
      v16bf kb10 = frag_b(&Ks[16][0], 72, lane);
      v16bf kb11 = frag_b(&Ks[16][32], 72, lane);
      s1 = wmma_bf16f32(qa0, kb10, s1);
      s1 = wmma_bf16f32(qa1, kb11, s1);
    }

    // online softmax update (row stats via half-wave shfl reductions)
    float p0[8], p1[8];
#pragma unroll
    for (int r = 0; r < 8; ++r) {
      const int trow = q0 + r + half * 8;
      float a  = s0[r] * 0.125f;
      float bq = s1[r] * 0.125f;
      if (key0 + cn > trow)      a  = -3.38953139e38f;
      if (key0 + 16 + cn > trow) bq = -3.38953139e38f;
      float mx = fmaxf(a, bq);
#pragma unroll
      for (int m = 8; m >= 1; m >>= 1) mx = fmaxf(mx, __shfl_xor(mx, m, 32));
      const float mnew  = fmaxf(mr[r], mx);
      const float scale = __expf(mr[r] - mnew);
      const float pa = __expf(a - mnew);
      const float pb = __expf(bq - mnew);
      float rs = pa + pb;
#pragma unroll
      for (int m = 8; m >= 1; m >>= 1) rs += __shfl_xor(rs, m, 32);
      lr[r] = lr[r] * scale + rs;
      mr[r] = mnew;
#pragma unroll
      for (int j = 0; j < 4; ++j) o[j][r] *= scale;
      p0[r] = pa;
      p1[r] = pb;
    }

    // stage P into A-frag layout (per-wave LDS ops are in-order on CDNA5)
#pragma unroll
    for (int r = 0; r < 8; ++r) {
      Pa[wave][half * 8 + r][cn]      = (__bf16)p0[r];
      Pa[wave][half * 8 + r][16 + cn] = (__bf16)p1[r];
    }
    const v16bf pf = frag_a(&Pa[wave][0][0], 40, lane);
#pragma unroll
    for (int j = 0; j < 4; ++j) {
      v16bf vbf = frag_b(&Vt[j * 16][0], 40, lane);
      o[j] = wmma_bf16f32(pf, vbf, o[j]);
    }
  }

  // finalize: O/l, write to y[B,T,C] bf16 (c = h*64 + d)
#pragma unroll
  for (int j = 0; j < 4; ++j) {
#pragma unroll
    for (int r = 0; r < 8; ++r) {
      const int trow = q0 + r + half * 8;
      const size_t n = (size_t)b * 2048 + trow;
      const int c = h * 64 + j * 16 + cn;
      Yb[n * 1024 + c] = (__bf16)(o[j][r] / lr[r]);
    }
  }
}

// ---------------- router: logits, softmax stats, argmax ----------------------
__global__ __launch_bounds__(256) void router_kernel(const float* __restrict__ h2f,
                                                     const float* __restrict__ Wr,
                                                     float* __restrict__ tokstat,
                                                     int* __restrict__ eid) {
  const int wave = threadIdx.x >> 5, lane = threadIdx.x & 31;
  const int n = blockIdx.x * 8 + wave;
  const float* hrow = h2f + (size_t)n * 1024;
  float a0 = 0.f, a1 = 0.f, a2 = 0.f, a3 = 0.f;
  for (int c = lane; c < 1024; c += 32) {
    const float v = hrow[c];
    const float* w = Wr + c * 4;
    a0 += v * w[0]; a1 += v * w[1]; a2 += v * w[2]; a3 += v * w[3];
  }
#pragma unroll
  for (int m = 16; m >= 1; m >>= 1) {
    a0 += __shfl_xor(a0, m, 32); a1 += __shfl_xor(a1, m, 32);
    a2 += __shfl_xor(a2, m, 32); a3 += __shfl_xor(a3, m, 32);
  }
  if (lane == 0) {
    float lg[4] = {a0, a1, a2, a3};
    int am = 0;
    for (int e2 = 1; e2 < 4; ++e2) if (lg[e2] > lg[am]) am = e2;  // first-max wins
    float mx = lg[0];
    for (int e2 = 1; e2 < 4; ++e2) mx = fmaxf(mx, lg[e2]);
    float p[4], s = 0.f;
    for (int e2 = 0; e2 < 4; ++e2) { p[e2] = __expf(lg[e2] - mx); s += p[e2]; }
    const float lse = mx + logf(s);
    float ent = 0.f;
    for (int e2 = 0; e2 < 4; ++e2) {
      const float pr = p[e2] / s;
      tokstat[(size_t)n * 8 + e2] = pr;
      ent -= pr * logf(pr + 1e-9f);
    }
    tokstat[(size_t)n * 8 + 4] = ent;
    tokstat[(size_t)n * 8 + 5] = lse * lse;
    eid[n] = am;
  }
}

// ---------------- deterministic stable partition by expert -------------------
__global__ void partition_kernel(const int* __restrict__ eid, int* __restrict__ perm,
                                 int* __restrict__ meta, float* __restrict__ countsf) {
  if (threadIdx.x != 0 || blockIdx.x != 0) return;
  int cnt[4] = {0, 0, 0, 0};
  for (int t = 0; t < 4096; ++t) cnt[eid[t]]++;
  int off[4];
  off[0] = 0;
  for (int e = 1; e < 4; ++e) off[e] = off[e - 1] + cnt[e - 1];
  for (int e = 0; e < 4; ++e) {
    meta[e] = cnt[e];
    meta[4 + e] = off[e];
    countsf[e] = (float)cnt[e];
  }
  int cur[4] = {off[0], off[1], off[2], off[3]};
  for (int t = 0; t < 4096; ++t) perm[cur[eid[t]]++] = t;  // stable, deterministic
}

// ---------------- fixed-tree stat reduction ----------------------------------
__global__ __launch_bounds__(256) void reduce_stats_kernel(const float* __restrict__ tokstat,
                                                           float* __restrict__ sums) {
  __shared__ float sh[256][6];
  float loc[6] = {0.f, 0.f, 0.f, 0.f, 0.f, 0.f};
  for (int t = threadIdx.x; t < 4096; t += 256)
#pragma unroll
    for (int k = 0; k < 6; ++k) loc[k] += tokstat[(size_t)t * 8 + k];
#pragma unroll
  for (int k = 0; k < 6; ++k) sh[threadIdx.x][k] = loc[k];
  __syncthreads();
  for (int s = 128; s > 0; s >>= 1) {
    if ((int)threadIdx.x < s)
#pragma unroll
      for (int k = 0; k < 6; ++k) sh[threadIdx.x][k] += sh[threadIdx.x + s][k];
    __syncthreads();
  }
  if (threadIdx.x == 0)
#pragma unroll
    for (int k = 0; k < 6; ++k) sums[k] = sh[0][k];
}

__global__ void finalize_kernel(const float* __restrict__ countsf,
                                const float* __restrict__ sums,
                                float* __restrict__ out_tail) {
  if (threadIdx.x != 0 || blockIdx.x != 0) return;
  const float total = 4096.0f;
  float aux = 0.f;
  for (int e = 0; e < 4; ++e) aux += (countsf[e] / total) * (sums[e] / total);
  out_tail[0] = 4.0f * aux;                 // aux
  out_tail[1] = sums[5] / total;            // z_loss
  out_tail[2] = (sums[4] / total) / logf(4.0f);  // router entropy
  for (int e = 0; e < 4; ++e) out_tail[3 + e] = countsf[e] / total;  // frac
}

// =============================== host launcher ===============================
extern "C" void kernel_launch(void* const* d_in, const int* in_sizes, int n_in,
                              void* d_out, int out_size, void* d_ws, size_t ws_size,
                              hipStream_t stream) {
  (void)in_sizes; (void)n_in; (void)out_size; (void)ws_size;
  const float* x   = (const float*)d_in[0];
  const float* Wq  = (const float*)d_in[1];
  const float* Wk  = (const float*)d_in[2];
  const float* Wv  = (const float*)d_in[3];
  const float* Wo  = (const float*)d_in[4];
  const float* Wr  = (const float*)d_in[5];
  const float* Wfc = (const float*)d_in[6];
  const float* Wpr = (const float*)d_in[7];
  float* out = (float*)d_out;

  const size_t NT = 4096, C = 1024, FF = 4096;

  char* base = (char*)d_ws;
  size_t off = 0;
  auto take = [&](size_t bytes) -> char* {
    char* cur = base + off;
    off = (off + bytes + 255) & ~(size_t)255;
    return cur;
  };

  __bf16* wq_b  = (__bf16*)take(C * C * 2);
  __bf16* wk_b  = (__bf16*)take(C * C * 2);
  __bf16* wv_b  = (__bf16*)take(C * C * 2);
  __bf16* wo_b  = (__bf16*)take(C * C * 2);
  __bf16* wfc_b = (__bf16*)take(4 * C * FF * 2);
  __bf16* wpr_b = (__bf16*)take(4 * FF * C * 2);
  __bf16* h1_b  = (__bf16*)take(NT * C * 2);
  float*  qf    = (float*)take(NT * C * 4);
  float*  kf    = (float*)take(NT * C * 4);
  float*  vf    = (float*)take(NT * C * 4);
  __bf16* Qh    = (__bf16*)take(NT * C * 2);
  __bf16* Kh    = (__bf16*)take(NT * C * 2);
  __bf16* Vh    = (__bf16*)take(NT * C * 2);
  __bf16* y_b   = (__bf16*)take(NT * C * 2);
  float*  x1    = (float*)take(NT * C * 4);
  __bf16* h2_b  = (__bf16*)take(NT * C * 2);
  float*  h2_f  = (float*)take(NT * C * 4);
  __bf16* hid_b = (__bf16*)take(NT * FF * 2);
  float*  tokst = (float*)take(NT * 8 * 4);
  int*    eid   = (int*)take(NT * 4);
  int*    perm  = (int*)take(NT * 4);
  int*    meta  = (int*)take(8 * 4);
  float*  cntf  = (float*)take(4 * 4);
  float*  sums  = (float*)take(6 * 4);

  // 1) weights -> bf16
  cast_bf16_kernel<<<4096, 256, 0, stream>>>(Wq, wq_b, C * C);
  cast_bf16_kernel<<<4096, 256, 0, stream>>>(Wk, wk_b, C * C);
  cast_bf16_kernel<<<4096, 256, 0, stream>>>(Wv, wv_b, C * C);
  cast_bf16_kernel<<<4096, 256, 0, stream>>>(Wo, wo_b, C * C);
  cast_bf16_kernel<<<4096, 256, 0, stream>>>(Wfc, wfc_b, 4 * C * FF);
  cast_bf16_kernel<<<4096, 256, 0, stream>>>(Wpr, wpr_b, 4 * FF * C);

  // 2) h1 = rmsnorm(x) -> bf16
  rmsnorm_kernel<false><<<NT, 256, 0, stream>>>(x, h1_b, nullptr);

  // 3) Q/K/V projections (WMMA)
  gemm_bf16_wmma<0><<<dim3(8, 32, 1), 256, 0, stream>>>(h1_b, wq_b, nullptr, qf, nullptr,
                                                        nullptr, nullptr, 4096, 1024, 1024);
  gemm_bf16_wmma<0><<<dim3(8, 32, 1), 256, 0, stream>>>(h1_b, wk_b, nullptr, kf, nullptr,
                                                        nullptr, nullptr, 4096, 1024, 1024);
  gemm_bf16_wmma<0><<<dim3(8, 32, 1), 256, 0, stream>>>(h1_b, wv_b, nullptr, vf, nullptr,
                                                        nullptr, nullptr, 4096, 1024, 1024);

  // 4) per-head RMSNorm + rotary -> [B,H,T,D] bf16
  qk_prep_kernel<<<8192, 256, 0, stream>>>(qf, kf, vf, Qh, Kh, Vh);

  // 5) causal flash attention
  flash_attn_kernel<<<512, 256, 0, stream>>>(Qh, Kh, Vh, y_b);

  // 6) x1 = x + y @ Wo (fused residual)
  gemm_bf16_wmma<1><<<dim3(8, 32, 1), 256, 0, stream>>>(y_b, wo_b, x, x1, nullptr,
                                                        nullptr, nullptr, 4096, 1024, 1024);

  // 7) h2 = rmsnorm(x1) (bf16 for GEMM, f32 for router)
  rmsnorm_kernel<true><<<NT, 256, 0, stream>>>(x1, h2_b, h2_f);

  // 8) router + stats, deterministic partition, reductions, scalar outputs
  router_kernel<<<512, 256, 0, stream>>>(h2_f, Wr, tokst, eid);
  partition_kernel<<<1, 32, 0, stream>>>(eid, perm, meta, cntf);
  reduce_stats_kernel<<<1, 256, 0, stream>>>(tokst, sums);
  finalize_kernel<<<1, 32, 0, stream>>>(cntf, sums, out + (size_t)NT * C);

  // 9) MoE expert GEMMs (gathered rows; z = expert)
  gemm_bf16_wmma<2><<<dim3(32, 32, 4), 256, 0, stream>>>(h2_b, wfc_b, nullptr, nullptr, hid_b,
                                                         perm, meta, 4096, 4096, 1024);
  gemm_bf16_wmma<3><<<dim3(8, 32, 4), 256, 0, stream>>>(hid_b, wpr_b, x1, out, nullptr,
                                                        perm, meta, 4096, 1024, 4096);
}
